// WiredCfC_8881992368277
// MI455X (gfx1250) — matrix operand order repaired and checked by reading, not compiled, not run on hardware
//
#include <hip/hip_runtime.h>
#include <math.h>

// ---------------------------------------------------------------------------
// WiredCfC persistent kernel for gfx1250 (MI455X).
//  - 1 workgroup per 16-row batch tile (32 WGs total), 256 threads = 8 waves.
//  - All fused weights (W1*mask, W2*mask, Wa+Wb) live in LDS as bf16 (~301KB).
//  - Hidden state lives in LDS concat buffers; the T=1024 scan runs inside
//    the kernel with only x_t loads and y_t stores touching global memory.
//  - GEMMs use v_wmma_f32_16x16x32_bf16 (M=16 batch rows, N=16 outs, K=32).
// ---------------------------------------------------------------------------

typedef __attribute__((ext_vector_type(16))) __bf16   v16bf;
typedef __attribute__((ext_vector_type(8)))  float    v8f;
typedef __attribute__((ext_vector_type(4)))  unsigned u32x4;

#define IN_DIM   64
#define H0       128
#define H1       96
#define H2       32
#define CAT0     (IN_DIM + H0)   // 192
#define CAT1     (H0 + H1)       // 224
#define CAT2     (H1 + H2)       // 128
#define BATCH    512
#define SEQ      1024
#define BT       16              // batch rows per workgroup
#define NTHREADS 256

union FragB16 {
    v16bf v;
    u32x4 q[2];
};

__device__ __forceinline__ __bf16 f2bf(float f) {
    union { float f; unsigned u; } c; c.f = f;
    unsigned r = (c.u + 0x7FFFu + ((c.u >> 16) & 1u)) >> 16;  // RNE
    union { unsigned short s; __bf16 b; } o; o.s = (unsigned short)r;
    return o.b;
}

// Fold mask into W1/W2, fuse Wa+Wb, convert to bf16 in LDS.
template<int H, int CAT>
__device__ __forceinline__ void prep_weights(__bf16* sW, float* sB,
                                             const float* const* in, int tid) {
    const float* W1 = in[0]; const float* b1 = in[1];
    const float* W2 = in[2]; const float* b2 = in[3];
    const float* Wa = in[4]; const float* ba = in[5];
    const float* Wb = in[6]; const float* bb = in[7];
    const float* mk = in[8];
    for (int i = tid; i < H * CAT; i += NTHREADS) {
        const float m = mk[i];
        sW[i]               = f2bf(W1[i] * m);
        sW[H * CAT + i]     = f2bf(W2[i] * m);
        sW[2 * H * CAT + i] = f2bf(Wa[i] + Wb[i]);
    }
    for (int i = tid; i < H; i += NTHREADS) {
        sB[i]         = b1[i];
        sB[H + i]     = b2[i];
        sB[2 * H + i] = ba[i] + bb[i];
    }
}

// One CfC layer: 3 WMMA GEMM chains + fused tanh/sigmoid interpolation.
// Each wave owns one 16-wide output tile (wave < H/16).
template<int CAT, int H>
__device__ __forceinline__ void cfc_compute(const __bf16* xc, const __bf16* W,
                                            const float* Bb, float hv[8], int tid) {
    const int wave = tid >> 5;
    const int lane = tid & 31;
    constexpr int NT = H / 16;
    if (wave < NT) {
        const int n0   = wave * 16;
        const int lo   = lane & 15;
        const int half = lane >> 4;                // 0 | 1
        // A fragment rows: 16 consecutive K per lane (lanes<16 K=0..15, lanes>=16 K=16..31)
        const __bf16* arow = xc + lo * CAT + (half << 4);
        // B fragment rows: 16 consecutive K per lane of W[n0+lo][...]
        const __bf16* w1 = W + (size_t)(n0 + lo) * CAT + (half << 4);
        const __bf16* w2 = w1 + (size_t)H * CAT;
        const __bf16* w3 = w2 + (size_t)H * CAT;

        v8f a1 = {}, a2 = {}, a3 = {};
#pragma unroll
        for (int k0 = 0; k0 < CAT; k0 += 32) {
            FragB16 a, bw1, bw2, bw3;
            a.q[0]   = *(const u32x4*)(arow + k0);
            a.q[1]   = *(const u32x4*)(arow + k0 + 8);
            bw1.q[0] = *(const u32x4*)(w1 + k0);
            bw1.q[1] = *(const u32x4*)(w1 + k0 + 8);
            bw2.q[0] = *(const u32x4*)(w2 + k0);
            bw2.q[1] = *(const u32x4*)(w2 + k0 + 8);
            bw3.q[0] = *(const u32x4*)(w3 + k0);
            bw3.q[1] = *(const u32x4*)(w3 + k0 + 8);
            a1 = __builtin_amdgcn_wmma_f32_16x16x32_bf16(false, a.v, false, bw1.v,
                                                         (short)0, a1, false, false);
            a2 = __builtin_amdgcn_wmma_f32_16x16x32_bf16(false, a.v, false, bw2.v,
                                                         (short)0, a2, false, false);
            a3 = __builtin_amdgcn_wmma_f32_16x16x32_bf16(false, a.v, false, bw3.v,
                                                         (short)0, a3, false, false);
        }

        const float b1v = Bb[0 * H + n0 + lo];
        const float b2v = Bb[1 * H + n0 + lo];
        const float b3v = Bb[2 * H + n0 + lo];
#pragma unroll
        for (int i = 0; i < 8; ++i) {
            const float f1 = tanhf(a1[i] + b1v);
            const float f2 = tanhf(a2[i] + b2v);
            const float s  = 1.0f / (1.0f + __expf(-(a3[i] + b3v)));
            hv[i] = f1 + s * (f2 - f1);           // ff1*(1-t) + t*ff2
        }
    }
}

// Scatter the C/D-layout tile into next-layer concat buffer, own h-region,
// and (layer 2 only) global output.
template<int H>
__device__ __forceinline__ void cfc_store(const float hv[8],
                                          __bf16* dA, int catA, int offA,
                                          __bf16* dB, int catB, int offB,
                                          float* gout, int goutStride, int tid) {
    const int wave = tid >> 5;
    const int lane = tid & 31;
    constexpr int NT = H / 16;
    if (wave < NT) {
        const int n0 = wave * 16;
        const int n  = lane & 15;
        const int mb = (lane >> 4) << 3;          // lanes>=16 -> M = i + 8
#pragma unroll
        for (int i = 0; i < 8; ++i) {
            const int m = mb + i;
            const __bf16 hb = f2bf(hv[i]);
            dA[m * catA + offA + n0 + n] = hb;
            if (dB)   dB[m * catB + offB + n0 + n] = hb;
            if (gout) gout[(size_t)m * goutStride + n0 + n] = hv[i];
        }
    }
}

struct KParams {
    const float* in[28];
    float*       out;
};

__global__ __launch_bounds__(NTHREADS, 1)
void wired_cfc_persistent(KParams P) {
    // ---- LDS: fused bf16 weights + concat/state buffers (~314 KB) ----
    __shared__ __align__(16) __bf16 sW0[3 * H0 * CAT0];   // 147456 B
    __shared__ __align__(16) __bf16 sW1[3 * H1 * CAT1];   // 129024 B
    __shared__ __align__(16) __bf16 sW2[3 * H2 * CAT2];   //  24576 B
    __shared__ __align__(16) __bf16 sXC0[BT * CAT0];      // [x_t | h0]
    __shared__ __align__(16) __bf16 sXC1[BT * CAT1];      // [h0  | h1]
    __shared__ __align__(16) __bf16 sXC2[BT * CAT2];      // [h1  | h2]
    __shared__ float sB0[3 * H0];
    __shared__ float sB1[3 * H1];
    __shared__ float sB2[3 * H2];

    const int tid = threadIdx.x;
    const int b0  = blockIdx.x * BT;

    prep_weights<H0, CAT0>(sW0, sB0, P.in + 1,  tid);
    prep_weights<H1, CAT1>(sW1, sB1, P.in + 10, tid);
    prep_weights<H2, CAT2>(sW2, sB2, P.in + 19, tid);

    // zero concat buffers (hidden state starts at 0)
    for (int i = tid; i < BT * CAT0; i += NTHREADS) ((unsigned short*)sXC0)[i] = 0;
    for (int i = tid; i < BT * CAT1; i += NTHREADS) ((unsigned short*)sXC1)[i] = 0;
    for (int i = tid; i < BT * CAT2; i += NTHREADS) ((unsigned short*)sXC2)[i] = 0;
    __syncthreads();

    const size_t xRowStride = (size_t)SEQ * IN_DIM;
    const float* xBase   = P.in[0] + (size_t)b0 * xRowStride;
    float*       outBase = P.out   + (size_t)b0 * SEQ * H2;

    for (int t = 0; t < SEQ; ++t) {
        // stage x_t (f32 -> bf16) into sXC0[:, 0:64]; prefetch x_{t+1}
        for (int i = tid; i < BT * IN_DIM; i += NTHREADS) {
            const int r = i >> 6, c = i & 63;
            const float* xp = xBase + (size_t)r * xRowStride + (size_t)t * IN_DIM + c;
            sXC0[r * CAT0 + c] = f2bf(*xp);
            if (((c & 15) == 0) && (t + 1 < SEQ))
                __builtin_prefetch(xp + IN_DIM, 0, 0);   // global_prefetch_b8
        }
        __syncthreads();

        float hv[8] = {0.f};

        // ---- layer 0: (16 x 192) @ (192 x 128) x3 ----
        cfc_compute<CAT0, H0>(sXC0, sW0, sB0, hv, tid);
        __syncthreads();
        cfc_store<H0>(hv, sXC1, CAT1, 0, sXC0, CAT0, IN_DIM, nullptr, 0, tid);
        __syncthreads();

        // ---- layer 1: (16 x 224) @ (224 x 96) x3 ----
        cfc_compute<CAT1, H1>(sXC1, sW1, sB1, hv, tid);
        __syncthreads();
        cfc_store<H1>(hv, sXC2, CAT2, 0, sXC1, CAT1, H0, nullptr, 0, tid);
        __syncthreads();

        // ---- layer 2: (16 x 128) @ (128 x 32) x3 ; emit y_t ----
        cfc_compute<CAT2, H2>(sXC2, sW2, sB2, hv, tid);
        __syncthreads();
        cfc_store<H2>(hv, sXC2, CAT2, H1, nullptr, 0, 0,
                      outBase + (size_t)t * H2, SEQ * H2, tid);
        __syncthreads();
    }
}

extern "C" void kernel_launch(void* const* d_in, const int* in_sizes, int n_in,
                              void* d_out, int out_size, void* d_ws, size_t ws_size,
                              hipStream_t stream) {
    (void)in_sizes; (void)out_size; (void)d_ws; (void)ws_size;
    KParams P;
    for (int i = 0; i < 28; ++i)
        P.in[i] = (i < n_in) ? (const float*)d_in[i] : nullptr;
    P.out = (float*)d_out;
    dim3 grid(BATCH / BT);     // 32 workgroups, one 16-row batch tile each
    dim3 block(NTHREADS);      // 8 wave32s
    hipLaunchKernelGGL(wired_cfc_persistent, grid, block, 0, stream, P);
}
